// DCNNv2_63118839382588
// MI455X (gfx1250) — compile-verified, matrix operand order checked by reading, not run on hardware
//
#include <hip/hip_runtime.h>
#include <hip/hip_bf16.h>
#include <math.h>

typedef __attribute__((ext_vector_type(16))) __bf16 v16bf;
typedef __attribute__((ext_vector_type(8)))  __bf16 v8bf;
typedef __attribute__((ext_vector_type(8)))  float  v8f;

#define DDIM   128
#define NEXT   10000
#define KNODES 16
#define JN     8
#define EN     8
#define NBATCH 1024
#define PITCH  136   // LDS row pitch (bf16 elems): 272B -> 4-bank rotation per row

__device__ __forceinline__ v16bf pack16(v8bf lo, v8bf hi) {
  v16bf r;
#pragma unroll
  for (int i = 0; i < 8; ++i) { r[i] = lo[i]; r[i + 8] = hi[i]; }
  return r;
}
__device__ __forceinline__ v16bf load16(const __bf16* p) {
  return pack16(*(const v8bf*)p, *(const v8bf*)(p + 16));
}

// ---------------- kernel 0: f32 -> bf16 weight conversion (W,M,U,V) ---------
__global__ __launch_bounds__(256) void k_cvt(
    const float* __restrict__ W, const float* __restrict__ M,
    const float* __restrict__ U, const float* __restrict__ V,
    __bf16* __restrict__ Wb, __bf16* __restrict__ Mb,
    __bf16* __restrict__ Ub, __bf16* __restrict__ Vb) {
  int i = blockIdx.x * 256 + threadIdx.x;   // 0 .. 16383
  Wb[i] = (__bf16)W[i];
  Mb[i] = (__bf16)M[i];
  Ub[i] = (__bf16)U[i];
  Vb[i] = (__bf16)V[i];
}

// ---------------- kernel 1: gather + WMMA (W,M) + relu + sum over K ---------
// one workgroup per node n; 8 waves, wave w owns output cols [16w,16w+16)
__global__ __launch_bounds__(256) void k_internal(
    const int*   __restrict__ nodes,    // (NEXT, 16)
    const int*   __restrict__ neigh,    // (NEXT, 16, 8)
    const float* __restrict__ Emb,      // (VOCAB, 128)
    const __bf16* __restrict__ Wb, const __bf16* __restrict__ Mb,
    float* __restrict__ int_pre)        // (NEXT, 128)  relu(s).sum(k)
{
  __shared__ __bf16 An[KNODES][PITCH];  // node embeddings, bf16
  __shared__ __bf16 Ag[KNODES][PITCH];  // neighbor sums,  bf16
  const int n  = blockIdx.x;
  const int t  = threadIdx.x;
  const int kk = t >> 4;            // 0..15 : k row
  const int cc = (t & 15) << 3;     // col chunk of 8

  { // node gather (16 rows x 128 cols, 8 f32 per thread)
    const int idx = nodes[n * KNODES + kk];
    const float4* s = (const float4*)(Emb + (size_t)idx * DDIM + cc);
    float4 a = s[0], b = s[1];
    __bf16* dst = &An[kk][cc];
    dst[0]=(__bf16)a.x; dst[1]=(__bf16)a.y; dst[2]=(__bf16)a.z; dst[3]=(__bf16)a.w;
    dst[4]=(__bf16)b.x; dst[5]=(__bf16)b.y; dst[6]=(__bf16)b.z; dst[7]=(__bf16)b.w;
  }
  { // neighbor gather + sum over J=8 (f32 accumulate, bf16 store)
    float acc[8] = {0.f,0.f,0.f,0.f,0.f,0.f,0.f,0.f};
#pragma unroll
    for (int j = 0; j < JN; ++j) {
      const int idx = neigh[(n * KNODES + kk) * JN + j];
      const float4* s = (const float4*)(Emb + (size_t)idx * DDIM + cc);
      float4 a = s[0], b = s[1];
      acc[0]+=a.x; acc[1]+=a.y; acc[2]+=a.z; acc[3]+=a.w;
      acc[4]+=b.x; acc[5]+=b.y; acc[6]+=b.z; acc[7]+=b.w;
    }
    __bf16* dst = &Ag[kk][cc];
#pragma unroll
    for (int i = 0; i < 8; ++i) dst[i] = (__bf16)acc[i];
  }
  __syncthreads();

  const int lane = t & 31;
  const int wv   = t >> 5;          // N-tile
  const int nr   = lane & 15;       // A: M row | B: N col (= output d within tile)
  const int hi   = lane >> 4;       // K-half select
  const __bf16* Wrow = Wb + (size_t)(wv * 16 + nr) * DDIM;  // B col n = W row d
  const __bf16* Mrow = Mb + (size_t)(wv * 16 + nr) * DDIM;

  v8f c = {};
#pragma unroll
  for (int ks = 0; ks < 4; ++ks) {
    const int e0 = ks * 32 + hi * 8;
    v16bf an = load16(&An[nr][e0]);
    v16bf ag = load16(&Ag[nr][e0]);
    v16bf bw = load16(Wrow + e0);
    v16bf bm = load16(Mrow + e0);
    c = __builtin_amdgcn_wmma_f32_16x16x32_bf16(false, an, false, bw, (short)0, c, false, false);
    c = __builtin_amdgcn_wmma_f32_16x16x32_bf16(false, ag, false, bm, (short)0, c, false, false);
  }
  // relu + sum over all 16 k-rows of this column
  float s = 0.f;
#pragma unroll
  for (int r = 0; r < 8; ++r) s += fmaxf(c[r], 0.f);
  s += __shfl_xor(s, 16, 32);       // partner lane holds other 8 rows, same column
  if (hi == 0) int_pre[(size_t)n * DDIM + wv * 16 + nr] = s;
}

// ---------------- softmax over rows of 128 (one wave per row) ---------------
__global__ __launch_bounds__(256) void k_softmax(
    const float* __restrict__ in, float* __restrict__ outf,
    __bf16* __restrict__ outb, int nrows)
{
  const int row  = (blockIdx.x * 256 + threadIdx.x) >> 5;
  const int lane = threadIdx.x & 31;
  if (row >= nrows) return;
  const float* r = in + (size_t)row * DDIM;
  float v[4]; float mx = -INFINITY;
#pragma unroll
  for (int i = 0; i < 4; ++i) { v[i] = r[lane + 32 * i]; mx = fmaxf(mx, v[i]); }
#pragma unroll
  for (int o = 16; o > 0; o >>= 1) mx = fmaxf(mx, __shfl_xor(mx, o, 32));
  float sum = 0.f;
#pragma unroll
  for (int i = 0; i < 4; ++i) { v[i] = __expf(v[i] - mx); sum += v[i]; }
#pragma unroll
  for (int o = 16; o > 0; o >>= 1) sum += __shfl_xor(sum, o, 32);
  const float inv = 1.f / sum;
#pragma unroll
  for (int i = 0; i < 4; ++i) {
    const float y = v[i] * inv;
    const size_t o = (size_t)row * DDIM + lane + 32 * i;
    if (outf) outf[o] = y;
    if (outb) outb[o] = (__bf16)y;
  }
}

// ---------------- neighbor sum of int_enc over E=8 -------------------------
__global__ __launch_bounds__(256) void k_nsum(
    const int* __restrict__ ext_neigh, const __bf16* __restrict__ enc,
    __bf16* __restrict__ nsum)
{
  const int gid = blockIdx.x * 256 + threadIdx.x;
  const int n = gid >> 7, d = gid & 127;
  if (n >= NEXT) return;
  float acc = 0.f;
#pragma unroll
  for (int i = 0; i < EN; ++i) {
    const int idx = ext_neigh[n * EN + i];
    acc += (float)enc[(size_t)idx * DDIM + d];
  }
  nsum[(size_t)n * DDIM + d] = (__bf16)acc;
}

// ---------------- kernel 4: WMMA (U,V) + relu ------------------------------
// one workgroup per 16 rows; same wave layout as k_internal
__global__ __launch_bounds__(256) void k_external(
    const __bf16* __restrict__ enc,   // (NEXT,128) bf16
    const __bf16* __restrict__ nsum,  // (NEXT,128) bf16
    const __bf16* __restrict__ Ub, const __bf16* __restrict__ Vb,
    float* __restrict__ ext_relu)     // (NEXT,128)
{
  __shared__ __bf16 A1[KNODES][PITCH];
  __shared__ __bf16 A2[KNODES][PITCH];
  const int rbase = blockIdx.x * 16;
  const int t   = threadIdx.x;
  const int row = t >> 4, col = (t & 15) << 3;
  *(v8bf*)&A1[row][col] = *(const v8bf*)(enc  + (size_t)(rbase + row) * DDIM + col);
  *(v8bf*)&A2[row][col] = *(const v8bf*)(nsum + (size_t)(rbase + row) * DDIM + col);
  __syncthreads();

  const int lane = t & 31;
  const int wv = t >> 5, nr = lane & 15, hi = lane >> 4;
  const __bf16* Urow = Ub + (size_t)(wv * 16 + nr) * DDIM;
  const __bf16* Vrow = Vb + (size_t)(wv * 16 + nr) * DDIM;

  v8f c = {};
#pragma unroll
  for (int ks = 0; ks < 4; ++ks) {
    const int e0 = ks * 32 + hi * 8;
    v16bf a1 = load16(&A1[nr][e0]);
    v16bf a2 = load16(&A2[nr][e0]);
    v16bf bu = load16(Urow + e0);
    v16bf bv = load16(Vrow + e0);
    c = __builtin_amdgcn_wmma_f32_16x16x32_bf16(false, a1, false, bu, (short)0, c, false, false);
    c = __builtin_amdgcn_wmma_f32_16x16x32_bf16(false, a2, false, bv, (short)0, c, false, false);
  }
#pragma unroll
  for (int r = 0; r < 8; ++r) {
    const int m = r + hi * 8;
    ext_relu[(size_t)(rbase + m) * DDIM + wv * 16 + nr] = fmaxf(c[r], 0.f);
  }
}

// ---------------- kernel 6: batch head ------------------------------------
__global__ __launch_bounds__(128) void k_head(
    const int* __restrict__ batch, const float* __restrict__ ext_enc,
    const float* __restrict__ W1, const float* __restrict__ b1,
    const float* __restrict__ W2, const float* __restrict__ b2,
    float* __restrict__ out)
{
  __shared__ float feat[2 * DDIM];
  __shared__ float r0[DDIM], r1[DDIM];
  const int b = blockIdx.x, d = threadIdx.x;
  const int i1 = batch[b * 2 + 0], i2 = batch[b * 2 + 1];
  const float a = ext_enc[(size_t)i1 * DDIM + d];
  const float c = ext_enc[(size_t)i2 * DDIM + d];
  feat[d] = a * c;
  feat[d + DDIM] = a + c;
  __syncthreads();
  float x = b1[d];
  const float* w1r = W1 + (size_t)d * 2 * DDIM;   // W1 is (D, 2D) row-major
#pragma unroll 8
  for (int j = 0; j < 2 * DDIM; ++j) x = fmaf(feat[j], w1r[j], x);
  r0[d] = x * W2[d];          // W2 row 0
  r1[d] = x * W2[DDIM + d];   // W2 row 1
  __syncthreads();
  for (int s = 64; s > 0; s >>= 1) {
    if (d < s) { r0[d] += r0[d + s]; r1[d] += r1[d + s]; }
    __syncthreads();
  }
  if (d == 0) {
    const float l0 = r0[0] + b2[0], l1 = r1[0] + b2[1];
    const float mx = fmaxf(l0, l1);
    const float e0 = __expf(l0 - mx), e1 = __expf(l1 - mx);
    const float inv = 1.f / (e0 + e1);
    out[b * 2 + 0] = e0 * inv;
    out[b * 2 + 1] = e1 * inv;
  }
}

extern "C" void kernel_launch(void* const* d_in, const int* in_sizes, int n_in,
                              void* d_out, int out_size, void* d_ws, size_t ws_size,
                              hipStream_t stream) {
  const int*   batch  = (const int*)d_in[0];
  const int*   nodes  = (const int*)d_in[1];
  const int*   neigh  = (const int*)d_in[2];
  const int*   eneigh = (const int*)d_in[3];
  const float* Emb    = (const float*)d_in[4];
  const float* W      = (const float*)d_in[5];
  const float* M      = (const float*)d_in[6];
  const float* U      = (const float*)d_in[7];
  const float* V      = (const float*)d_in[8];
  const float* W1     = (const float*)d_in[9];
  const float* b1     = (const float*)d_in[10];
  const float* W2     = (const float*)d_in[11];
  const float* b2     = (const float*)d_in[12];
  float* out = (float*)d_out;

  // workspace carve-up (~20.6 MB)
  char* base = (char*)d_ws;
  __bf16* Wb = (__bf16*)base;                       // 16384 elems
  __bf16* Mb = Wb + DDIM * DDIM;
  __bf16* Ub = Mb + DDIM * DDIM;
  __bf16* Vb = Ub + DDIM * DDIM;                    // ends at 131072 B
  float*  int_pre  = (float*)(base + 4 * DDIM * DDIM * sizeof(__bf16));
  __bf16* int_enc  = (__bf16*)((char*)int_pre + (size_t)NEXT * DDIM * sizeof(float));
  __bf16* nsum     = int_enc + (size_t)NEXT * DDIM;
  float*  ext_relu = (float*)((char*)(nsum + (size_t)NEXT * DDIM));
  float*  ext_enc  = ext_relu + (size_t)NEXT * DDIM;

  k_cvt<<<(DDIM * DDIM) / 256, 256, 0, stream>>>(W, M, U, V, Wb, Mb, Ub, Vb);
  k_internal<<<NEXT, 256, 0, stream>>>(nodes, neigh, Emb, Wb, Mb, int_pre);
  k_softmax<<<(NEXT + 7) / 8, 256, 0, stream>>>(int_pre, nullptr, int_enc, NEXT);
  k_nsum<<<(NEXT * DDIM) / 256, 256, 0, stream>>>(eneigh, int_enc, nsum);
  k_external<<<NEXT / 16, 256, 0, stream>>>(int_enc, nsum, Ub, Vb, ext_relu);
  k_softmax<<<(NEXT + 7) / 8, 256, 0, stream>>>(ext_relu, ext_enc, nullptr, NEXT);
  k_head<<<NBATCH, DDIM, 0, stream>>>(batch, ext_enc, W1, b1, W2, b2, out);
}
